// Block_70720931496317
// MI455X (gfx1250) — compile-verified, matrix operand order checked by reading, not run on hardware
//
#include <hip/hip_runtime.h>
#include <cstdint>
#include <cstddef>

// Problem constants (match reference)
constexpr int Bc = 2, Sc = 2048, Dc = 1024, Hc = 16, DHc = 64;
constexpr int Mrows = Bc * Sc;           // 4096
constexpr float EPSc = 1e-5f;

#define USE_ASYNC_COPY 1
#if defined(__has_builtin)
#if __has_builtin(__builtin_amdgcn_tensor_load_to_lds)
#define USE_TDM 1
#else
#define USE_TDM 0
#endif
#else
#define USE_TDM 0
#endif

typedef __attribute__((ext_vector_type(16))) __bf16 v16bf;
typedef __attribute__((ext_vector_type(8)))  float  v8f;

union Frag {
  v16bf v;
  struct { uint4 lo, hi; } q;
  unsigned short u[16];
};

__device__ __forceinline__ unsigned short f2bf(float f) {
  union { float f; unsigned int u; } c; c.f = f;
  unsigned int u = c.u;
  return (unsigned short)((u + 0x7fffu + ((u >> 16) & 1u)) >> 16);
}

#if USE_ASYNC_COPY
__device__ __forceinline__ void async_copy_b128(unsigned lds_off, const unsigned short* g) {
  asm volatile("global_load_async_to_lds_b128 %0, %1, off"
               :: "v"(lds_off), "v"(g) : "memory");
}
__device__ __forceinline__ void wait_async() {
  asm volatile("s_wait_asynccnt 0x0" ::: "memory");
}
#endif

#if USE_TDM
typedef __attribute__((ext_vector_type(4))) unsigned int u32x4;
typedef __attribute__((ext_vector_type(8))) int         i32x8;
typedef __attribute__((ext_vector_type(4))) int         i32x4;

__device__ __forceinline__ void wait_tensor() {
#if __has_builtin(__builtin_amdgcn_s_wait_tensorcnt)
  __builtin_amdgcn_s_wait_tensorcnt(0);
#else
  asm volatile("s_wait_tensorcnt 0x0" ::: "memory");
#endif
}

// DMA one 32(k) x 128(row) bf16 tile into LDS with 16B padding per 64B row
// (LDS row stride 80B = [row][40] ushort layout used by the fragment loads).
template<int K>
__device__ __forceinline__ void tdm_tile_load(unsigned lds_off, const unsigned short* gaddr) {
  unsigned long long ga = (unsigned long long)(size_t)gaddr;
  u32x4 g0;
  g0[0] = 1u;                                  // count=1, is_restore=0, gather off
  g0[1] = lds_off;                             // D# lds_addr
  g0[2] = (unsigned)(ga & 0xffffffffu);        // global_addr[31:0]
  g0[3] = (unsigned)(ga >> 32) | (2u << 30);   // global_addr[56:32] | type=2 (image)
  i32x8 g1;
  g1[0] = (int)((1u << 16)    // data_size = 2 bytes
              | (1u << 20)    // pad_enable
              | (3u << 22)    // pad_interval: 16 dwords (64B) between pads
              | (3u << 25));  // pad_amount: 4 dwords (16B)
  g1[1] = (int)(((unsigned)K & 0xffffu) << 16);          // tensor_dim0 lo16
  g1[2] = (int)(((unsigned)K >> 16) | (128u << 16));     // tensor_dim0 hi16 | tensor_dim1 lo16
  g1[3] = (int)(32u << 16);                              // tensor_dim1 hi16 | tile_dim0=32
  g1[4] = (int)128u;                                     // tile_dim1=128, tile_dim2=0
  g1[5] = (int)K;                                        // tensor_dim0_stride lo32
  g1[6] = 0;                                             // stride0 hi16 | stride1 lo16
  g1[7] = 0;
  i32x4 z = {0, 0, 0, 0};
#if __clang_major__ >= 23
  i32x8 z8 = {0, 0, 0, 0, 0, 0, 0, 0};
  __builtin_amdgcn_tensor_load_to_lds(g0, g1, z, z, z8, 0);
#else
  __builtin_amdgcn_tensor_load_to_lds(g0, g1, z, z, 0);
#endif
}
#endif  // USE_TDM

// ---------------- weight conversion fp32 -> bf16, transposed (W[k][n] -> Wt[n][k]) ----------------
__global__ void cvt_bf16_t(const float* __restrict__ src, unsigned short* __restrict__ dst,
                           int K, int N) {
  int i = blockIdx.x * blockDim.x + threadIdx.x;   // over K*N/4
  if (i >= (K * N) / 4) return;
  int e = i * 4;
  int k = e / N, n = e % N;
  float4 v = *(const float4*)(src + (size_t)k * N + n);
  dst[(size_t)(n + 0) * K + k] = f2bf(v.x);
  dst[(size_t)(n + 1) * K + k] = f2bf(v.y);
  dst[(size_t)(n + 2) * K + k] = f2bf(v.z);
  dst[(size_t)(n + 3) * K + k] = f2bf(v.w);
}

__global__ void mask_x(const float* __restrict__ x,
                       const unsigned char* __restrict__ mask,
                       float* __restrict__ xf, unsigned short* __restrict__ xb) {
  int i = blockIdx.x * blockDim.x + threadIdx.x;   // over Mrows*Dc/4
  if (i >= Mrows * Dc / 4) return;
  int r = (i * 4) >> 10;
  float m = mask[r] ? 1.0f : 0.0f;
  float4 v = *(const float4*)(x + (size_t)i * 4);
  v.x *= m; v.y *= m; v.z *= m; v.w *= m;
  *(float4*)(xf + (size_t)i * 4) = v;
  ushort4 o;
  o.x = f2bf(v.x); o.y = f2bf(v.y); o.z = f2bf(v.z); o.w = f2bf(v.w);
  *(ushort4*)(xb + (size_t)i * 4) = o;
}

// ---------------- bf16 WMMA GEMM: C = A(MxK) * Bt(NxK)^T + bias ----------------
template<int N, int K, bool WF, bool WB16, bool RELU>
__launch_bounds__(256)
__global__ void gemm_bf16(const unsigned short* __restrict__ A,
                          const unsigned short* __restrict__ Bt,
                          const float* __restrict__ bias,
                          float* __restrict__ Cf,
                          unsigned short* __restrict__ Cb)
{
  __shared__ unsigned short As[2][128 * 40];   // [row][k], 16B pad per row
  __shared__ unsigned short Bs[2][128 * 40];   // [n][k],   16B pad per row
  const int tid  = threadIdx.x;
  const int wave = tid >> 5, l = tid & 31;
  const int wm = wave >> 1, wn = wave & 1;
  const int half = l >> 4, lm = l & 15;
  const int bm0 = blockIdx.y * 128, bn0 = blockIdx.x * 128;

  const int r0 = tid >> 2,         c0 = (tid & 3) * 8;
  const int r1 = (tid + 256) >> 2, c1 = ((tid + 256) & 3) * 8;

  const unsigned short* Abase = A  + (size_t)bm0 * K;
  const unsigned short* Bbase = Bt + (size_t)bn0 * K;

  auto issue_tile = [&](int buf, int kb) {
#if USE_TDM
    if (wave == 0) {
      tdm_tile_load<K>((unsigned)(size_t)&As[buf][0], Abase + kb * 32);
      tdm_tile_load<K>((unsigned)(size_t)&Bs[buf][0], Bbase + kb * 32);
    }
#elif USE_ASYNC_COPY
    const unsigned short* Ak = Abase + kb * 32;
    const unsigned short* Bk = Bbase + kb * 32;
    async_copy_b128((unsigned)(size_t)&As[buf][r0 * 40 + c0], Ak + (size_t)r0 * K + c0);
    async_copy_b128((unsigned)(size_t)&As[buf][r1 * 40 + c1], Ak + (size_t)r1 * K + c1);
    async_copy_b128((unsigned)(size_t)&Bs[buf][r0 * 40 + c0], Bk + (size_t)r0 * K + c0);
    async_copy_b128((unsigned)(size_t)&Bs[buf][r1 * 40 + c1], Bk + (size_t)r1 * K + c1);
#else
    const unsigned short* Ak = Abase + kb * 32;
    const unsigned short* Bk = Bbase + kb * 32;
    *(uint4*)&As[buf][r0 * 40 + c0] = *(const uint4*)(Ak + (size_t)r0 * K + c0);
    *(uint4*)&As[buf][r1 * 40 + c1] = *(const uint4*)(Ak + (size_t)r1 * K + c1);
    *(uint4*)&Bs[buf][r0 * 40 + c0] = *(const uint4*)(Bk + (size_t)r0 * K + c0);
    *(uint4*)&Bs[buf][r1 * 40 + c1] = *(const uint4*)(Bk + (size_t)r1 * K + c1);
#endif
  };

  auto wait_tiles = [&]() {
#if USE_TDM
    if (wave == 0) wait_tensor();
#elif USE_ASYNC_COPY
    wait_async();
#endif
    __syncthreads();
  };

  issue_tile(0, 0);
  wait_tiles();

  v8f acc[8];
#pragma unroll
  for (int i = 0; i < 8; ++i) acc[i] = {};

  constexpr int kblocks = K >> 5;
  for (int kb = 0; kb < kblocks; ++kb) {
    const int cur = kb & 1;
    if (kb + 1 < kblocks) {
      issue_tile(cur ^ 1, kb + 1);
      if (kb + 2 < kblocks) {   // L2 prefetch two tiles ahead
        __builtin_prefetch(Abase + (kb + 2) * 32 + (size_t)r0 * K + c0, 0, 1);
        __builtin_prefetch(Bbase + (kb + 2) * 32 + (size_t)r0 * K + c0, 0, 1);
      }
    }

#pragma unroll
    for (int tm = 0; tm < 2; ++tm) {
      Frag af;
      const unsigned short* ap = As[cur] + (wm * 32 + tm * 16 + lm) * 40 + half * 8;
      af.q.lo = *(const uint4*)ap;
      af.q.hi = *(const uint4*)(ap + 16);
#pragma unroll
      for (int tn = 0; tn < 4; ++tn) {
        Frag bf;
        const unsigned short* bp = Bs[cur] + (wn * 64 + tn * 16 + lm) * 40 + half * 8;
        bf.q.lo = *(const uint4*)bp;
        bf.q.hi = *(const uint4*)(bp + 16);
        acc[tm * 4 + tn] = __builtin_amdgcn_wmma_f32_16x16x32_bf16(
            false, af.v, false, bf.v, (short)0, acc[tm * 4 + tn], false, false);
      }
    }

    if (kb + 1 < kblocks) wait_tiles();
  }

  // epilogue: branch-free, compile-time layout
#pragma unroll
  for (int tm = 0; tm < 2; ++tm) {
    const int row0 = bm0 + wm * 32 + tm * 16 + half * 8;
#pragma unroll
    for (int tn = 0; tn < 4; ++tn) {
      const int c = bn0 + wn * 64 + tn * 16 + lm;
      const float bv = bias[c];
      float* cf = WF ? (Cf + (size_t)row0 * N + c) : nullptr;
      unsigned short* cb = WB16 ? (Cb + (size_t)row0 * N + c) : nullptr;
#pragma unroll
      for (int i = 0; i < 8; ++i) {
        float v = acc[tm * 4 + tn][i] + bv;
        if (RELU) v = v > 0.0f ? v : 0.0f;
        if (WF)   cf[(size_t)i * N] = v;
        if (WB16) cb[(size_t)i * N] = f2bf(v);
      }
    }
  }
}

// ---------------- split qkv fp32 -> q,k bf16 [B,H,S,DH]; v transposed [B,H,DH,S] ----------------
__global__ void split_qkv(const float* __restrict__ qkv,
                          unsigned short* __restrict__ q,
                          unsigned short* __restrict__ k,
                          unsigned short* __restrict__ vt) {
  int i = blockIdx.x * blockDim.x + threadIdx.x;    // over Mrows*Dc/4
  if (i >= Mrows * Dc / 4) return;
  int e = i * 4;
  int r = e >> 10, d = e & 1023;
  int h = d >> 6, dh = d & 63;
  int b = r >> 11, s = r & 2047;
  size_t base = (size_t)r * 3072;
  int bh = b * Hc + h;
  float4 vq = *(const float4*)(qkv + base + d);
  float4 vk = *(const float4*)(qkv + base + 1024 + d);
  float4 vv = *(const float4*)(qkv + base + 2048 + d);
  ushort4 oq; oq.x = f2bf(vq.x); oq.y = f2bf(vq.y); oq.z = f2bf(vq.z); oq.w = f2bf(vq.w);
  ushort4 ok; ok.x = f2bf(vk.x); ok.y = f2bf(vk.y); ok.z = f2bf(vk.z); ok.w = f2bf(vk.w);
  *(ushort4*)(q + ((size_t)bh * Sc + s) * 64 + dh) = oq;
  *(ushort4*)(k + ((size_t)bh * Sc + s) * 64 + dh) = ok;
  unsigned short* vtp = vt + (size_t)bh * 64 * Sc + (size_t)dh * Sc + s;
  vtp[0 * Sc] = f2bf(vv.x);
  vtp[1 * Sc] = f2bf(vv.y);
  vtp[2 * Sc] = f2bf(vv.z);
  vtp[3 * Sc] = f2bf(vv.w);
}

// ---------------- flash attention (causal + pad mask), bf16 WMMA ----------------
__launch_bounds__(256)
__global__ void attn(const unsigned short* __restrict__ q,
                     const unsigned short* __restrict__ k,
                     const unsigned short* __restrict__ vt,
                     const unsigned char* __restrict__ mask,
                     unsigned short* __restrict__ vw)
{
  __shared__ unsigned short pl[8 * 16 * 40];   // per-wave 16x32 P tile, stride 40
  const int tid = threadIdx.x, wave = tid >> 5, l = tid & 31;
  const int half = l >> 4, lm = l & 15;
  const int bh = blockIdx.y, b = bh / Hc, h = bh % Hc;
  const int qb = blockIdx.x * 128 + wave * 16;

  const unsigned short* qp = q  + (size_t)bh * Sc * 64;
  const unsigned short* kp = k  + (size_t)bh * Sc * 64;
  const unsigned short* vp = vt + (size_t)bh * 64 * Sc;
  const unsigned char*  mp = mask + (size_t)b * Sc;

  Frag qf[2];
#pragma unroll
  for (int c = 0; c < 2; ++c) {
    const unsigned short* p = qp + (size_t)(qb + lm) * 64 + c * 32 + half * 8;
    qf[c].q.lo = *(const uint4*)p;
    qf[c].q.hi = *(const uint4*)(p + 16);
  }

  v8f o[4];
#pragma unroll
  for (int t = 0; t < 4; ++t) o[t] = {};
  float mi[8], li[8];
#pragma unroll
  for (int i = 0; i < 8; ++i) { mi[i] = -1e30f; li[i] = 0.0f; }

  const int nblk = (qb + 47) >> 5;
  for (int kbI = 0; kbI < nblk; ++kbI) {
    const int key0 = kbI * 32;
    v8f s[2];
#pragma unroll
    for (int t = 0; t < 2; ++t) {
      Frag kf0, kf1;
      const unsigned short* p0 = kp + (size_t)(key0 + t * 16 + lm) * 64 + half * 8;
      kf0.q.lo = *(const uint4*)p0;        kf0.q.hi = *(const uint4*)(p0 + 16);
      kf1.q.lo = *(const uint4*)(p0 + 32); kf1.q.hi = *(const uint4*)(p0 + 48);
      v8f z = {};
      z = __builtin_amdgcn_wmma_f32_16x16x32_bf16(false, qf[0].v, false, kf0.v, (short)0, z, false, false);
      z = __builtin_amdgcn_wmma_f32_16x16x32_bf16(false, qf[1].v, false, kf1.v, (short)0, z, false, false);
      s[t] = z;
    }

    const int keyA = key0 + lm, keyB = key0 + 16 + lm;
    const float mA = mp[keyA] ? 0.0f : -1e30f;
    const float mB = mp[keyB] ? 0.0f : -1e30f;
#pragma unroll
    for (int i = 0; i < 8; ++i) {
      int row = qb + half * 8 + i;
      float s0 = s[0][i] * 0.125f + mA; if (keyA > row) s0 = -1e30f;
      float s1 = s[1][i] * 0.125f + mB; if (keyB > row) s1 = -1e30f;
      float r = fmaxf(s0, s1);
      r = fmaxf(r, __shfl_xor(r, 1, 32));
      r = fmaxf(r, __shfl_xor(r, 2, 32));
      r = fmaxf(r, __shfl_xor(r, 4, 32));
      r = fmaxf(r, __shfl_xor(r, 8, 32));
      float nm = fmaxf(mi[i], r);
      float alpha = __expf(mi[i] - nm);
      float p0v = __expf(s0 - nm), p1v = __expf(s1 - nm);
      float rs = p0v + p1v;
      rs += __shfl_xor(rs, 1, 32);
      rs += __shfl_xor(rs, 2, 32);
      rs += __shfl_xor(rs, 4, 32);
      rs += __shfl_xor(rs, 8, 32);
      li[i] = li[i] * alpha + rs;
      mi[i] = nm;
#pragma unroll
      for (int t = 0; t < 4; ++t) o[t][i] *= alpha;
      int prow = wave * 640 + (half * 8 + i) * 40;
      pl[prow + lm]      = f2bf(p0v);
      pl[prow + 16 + lm] = f2bf(p1v);
    }
    asm volatile("s_wait_dscnt 0" ::: "memory");

    Frag pf;
    const unsigned short* pp = pl + wave * 640 + lm * 40 + half * 8;
    pf.q.lo = *(const uint4*)pp;
    pf.q.hi = *(const uint4*)(pp + 16);
#pragma unroll
    for (int t = 0; t < 4; ++t) {
      Frag vf;
      const unsigned short* vpp = vp + (size_t)(t * 16 + lm) * Sc + key0 + half * 8;
      vf.q.lo = *(const uint4*)vpp;
      vf.q.hi = *(const uint4*)(vpp + 16);
      o[t] = __builtin_amdgcn_wmma_f32_16x16x32_bf16(false, pf.v, false, vf.v, (short)0, o[t], false, false);
    }
  }

#pragma unroll
  for (int i = 0; i < 8; ++i) {
    float inv = 1.0f / li[i];
    int r = b * Sc + qb + half * 8 + i;
#pragma unroll
    for (int t = 0; t < 4; ++t)
      vw[(size_t)r * Dc + h * 64 + t * 16 + lm] = f2bf(o[t][i] * inv);
  }
}

// ---------------- fused residual add + LayerNorm (one block per row) ----------------
__launch_bounds__(256)
__global__ void add_ln(const float* __restrict__ X, const float* __restrict__ Y,
                       const float* __restrict__ g, const float* __restrict__ bta,
                       float* __restrict__ of, unsigned short* __restrict__ ob)
{
  __shared__ float red[256];
  const int r = blockIdx.x, tid = threadIdx.x;
  float v[4];
  float s = 0.0f;
#pragma unroll
  for (int j = 0; j < 4; ++j) {
    int c = tid + j * 256;
    v[j] = X[(size_t)r * Dc + c] + Y[(size_t)r * Dc + c];
    s += v[j];
  }
  red[tid] = s; __syncthreads();
  for (int st = 128; st > 0; st >>= 1) { if (tid < st) red[tid] += red[tid + st]; __syncthreads(); }
  float u = red[0] / Dc;
  __syncthreads();
  float s2 = 0.0f;
#pragma unroll
  for (int j = 0; j < 4; ++j) { float d = v[j] - u; s2 += d * d; }
  red[tid] = s2; __syncthreads();
  for (int st = 128; st > 0; st >>= 1) { if (tid < st) red[tid] += red[tid + st]; __syncthreads(); }
  float rstd = rsqrtf(red[0] / Dc + EPSc);
#pragma unroll
  for (int j = 0; j < 4; ++j) {
    int c = tid + j * 256;
    float n = g[c] * ((v[j] - u) * rstd) + bta[c];
    of[(size_t)r * Dc + c] = n;
    if (ob) ob[(size_t)r * Dc + c] = f2bf(n);
  }
}

// ---------------- host orchestration ----------------
extern "C" void kernel_launch(void* const* d_in, const int* in_sizes, int n_in,
                              void* d_out, int out_size, void* d_ws, size_t ws_size,
                              hipStream_t stream) {
  const float*         x    = (const float*)d_in[0];
  const unsigned char* mask = (const unsigned char*)d_in[1];
  const float* Wqkv = (const float*)d_in[2];
  const float* bqkv = (const float*)d_in[3];
  const float* Wvw  = (const float*)d_in[4];
  const float* bvw  = (const float*)d_in[5];
  const float* g1   = (const float*)d_in[6];
  const float* b1   = (const float*)d_in[7];
  const float* WA   = (const float*)d_in[8];
  const float* bA   = (const float*)d_in[9];
  const float* WB   = (const float*)d_in[10];
  const float* bB   = (const float*)d_in[11];
  const float* g2   = (const float*)d_in[12];
  const float* b2   = (const float*)d_in[13];
  float* out = (float*)d_out;

  char* w = (char*)d_ws;
  auto alloc = [&](size_t bytes) -> void* {
    void* p = (void*)w;
    w += (bytes + 255) & ~((size_t)255);
    return p;
  };

  unsigned short* wqkv_b = (unsigned short*)alloc((size_t)Dc * 3 * Dc * 2);  // [3D][D]
  unsigned short* wvw_b  = (unsigned short*)alloc((size_t)Dc * Dc * 2);      // [D][D]
  unsigned short* wA_b   = (unsigned short*)alloc((size_t)Dc * 4 * Dc * 2);  // [4D][D]
  unsigned short* wB_b   = (unsigned short*)alloc((size_t)4 * Dc * Dc * 2);  // [D][4D]
  float*          xm_f   = (float*)alloc((size_t)Mrows * Dc * 4);
  unsigned short* xm_b   = (unsigned short*)alloc((size_t)Mrows * Dc * 2);
  float*          qkv_f  = (float*)alloc((size_t)Mrows * 3 * Dc * 4);
  unsigned short* q_b    = (unsigned short*)alloc((size_t)Mrows * Dc * 2);
  unsigned short* k_b    = (unsigned short*)alloc((size_t)Mrows * Dc * 2);
  unsigned short* vt_b   = (unsigned short*)alloc((size_t)Mrows * Dc * 2);
  unsigned short* vw_b   = (unsigned short*)alloc((size_t)Mrows * Dc * 2);
  float*          a_f    = (float*)alloc((size_t)Mrows * Dc * 4);
  float*          n_f    = (float*)alloc((size_t)Mrows * Dc * 4);
  unsigned short* n_b    = (unsigned short*)alloc((size_t)Mrows * Dc * 2);
  unsigned short* t_b    = (unsigned short*)alloc((size_t)Mrows * 4 * Dc * 2);
  float*          m_f    = (float*)alloc((size_t)Mrows * Dc * 4);

  const int ew = 256;
  // 1. weight conversions, transposed (Wt[n][k])
  cvt_bf16_t<<<(Dc * 3 * Dc / 4 + ew - 1) / ew, ew, 0, stream>>>(Wqkv, wqkv_b, Dc, 3 * Dc);
  cvt_bf16_t<<<(Dc * Dc / 4 + ew - 1) / ew, ew, 0, stream>>>(Wvw, wvw_b, Dc, Dc);
  cvt_bf16_t<<<(Dc * 4 * Dc / 4 + ew - 1) / ew, ew, 0, stream>>>(WA, wA_b, Dc, 4 * Dc);
  cvt_bf16_t<<<(4 * Dc * Dc / 4 + ew - 1) / ew, ew, 0, stream>>>(WB, wB_b, 4 * Dc, Dc);
  // 2. mask input
  mask_x<<<(Mrows * Dc / 4) / ew, ew, 0, stream>>>(x, mask, xm_f, xm_b);
  // 3. QKV projection
  gemm_bf16<3 * Dc, Dc, true, false, false><<<dim3(3 * Dc / 128, Mrows / 128), 256, 0, stream>>>(
      xm_b, wqkv_b, bqkv, qkv_f, nullptr);
  // 4. split + transpose V
  split_qkv<<<(Mrows * Dc / 4) / ew, ew, 0, stream>>>(qkv_f, q_b, k_b, vt_b);
  // 5. attention
  attn<<<dim3(Sc / 128, Bc * Hc), 256, 0, stream>>>(q_b, k_b, vt_b, mask, vw_b);
  // 6. output projection
  gemm_bf16<Dc, Dc, true, false, false><<<dim3(Dc / 128, Mrows / 128), 256, 0, stream>>>(
      vw_b, wvw_b, bvw, a_f, nullptr);
  // 7. n = LN(xm + a)
  add_ln<<<Mrows, 256, 0, stream>>>(xm_f, a_f, g1, b1, n_f, n_b);
  // 8. t = relu(n @ WA + bA)   (bf16 only)
  gemm_bf16<4 * Dc, Dc, false, true, true><<<dim3(4 * Dc / 128, Mrows / 128), 256, 0, stream>>>(
      n_b, wA_b, bA, nullptr, t_b);
  // 9. m = t @ WB + bB
  gemm_bf16<Dc, 4 * Dc, true, false, false><<<dim3(Dc / 128, Mrows / 128), 256, 0, stream>>>(
      t_b, wB_b, bB, m_f, nullptr);
  // 10. out = LN(n + m)
  add_ln<<<Mrows, 256, 0, stream>>>(n_f, m_f, g2, b2, out, nullptr);
}
